// ComplexAttention_41669772706530
// MI455X (gfx1250) — compile-verified, matrix operand order checked by reading, not run on hardware
//
#include <hip/hip_runtime.h>
#include <math.h>

// ---------------- types ----------------
typedef __attribute__((ext_vector_type(16))) __bf16 v16bf;
typedef __attribute__((ext_vector_type(8)))  float  v8f;
typedef __attribute__((ext_vector_type(8)))  __bf16 bf16x8;

#define BMD 128   // block tile M
#define BND 128   // block tile N
#define BKD 32    // block tile K (one wmma K-step)
#define BKP 34    // padded LDS row stride (odd dword count -> conflict-free gathers)

// ---------------- fp32 -> bf16 convert ----------------
__global__ void f32_to_bf16_kernel(const float* __restrict__ in,
                                   __bf16* __restrict__ out, long long n) {
  long long i = (long long)blockIdx.x * blockDim.x + threadIdx.x;
  long long stride = (long long)gridDim.x * blockDim.x;
  for (; i < n; i += stride) out[i] = (__bf16)in[i];
}

// ---------------- WMMA fragment gather from LDS ----------------
// Fragment rows r0..r0+15 of an LDS tile stored [row][K] with row stride BKP.
// CDNA5 16-bit A/B layout: lane&15 = row, lane>>4 = K half,
// VGPR v holds K pair ((v<4 ? 2v : 2v+8) + 8*half, +1).
static __device__ __forceinline__ v16bf ld_frag(const __bf16* lds, int r0, int lane) {
  const int l15  = lane & 15;
  const int half = (lane >> 4) & 1;
  union { v16bf v; unsigned u[8]; } f;
  const __bf16* p = lds + (unsigned)(r0 + l15) * BKP;
#pragma unroll
  for (int vv = 0; vv < 8; ++vv) {
    const int kb = ((vv < 4) ? (2 * vv) : (2 * vv + 8)) + 8 * half;  // even -> b32 load
    f.u[vv] = *(const unsigned*)(p + kb);
  }
  return f.v;
}

// ---------------- async global -> LDS tile copy (128 rows x 32 bf16) ----------------
// Uses gfx1250 GLOBAL_LOAD_ASYNC_TO_LDS_B128 (ASYNCcnt-tracked, no VGPR round-trip).
static __device__ __forceinline__ void async_tile_128x32(
    const __bf16* __restrict__ src, long long ld, int row0, int kt,
    __bf16 (*dst)[BKP], int tid)
{
#pragma unroll
  for (int i = 0; i < 2; ++i) {
    const int id  = tid + 256 * i;
    const int row = id >> 2;          // 4 x b128 per 32-elem row
    const int c8  = (id & 3) * 8;
    const __bf16* g = src + (long long)(row0 + row) * ld + (kt + c8);
    const unsigned loff = (unsigned)(size_t)&dst[row][c8];  // DS-relative byte offset
    asm volatile("global_load_async_to_lds_b128 %0, %1, off"
                 :: "v"(loff), "v"(g) : "memory");
  }
}

static __device__ __forceinline__ void async_wait0() {
  asm volatile("s_wait_asynccnt 0x0" ::: "memory");
}

// ---------------- generic bf16 WMMA GEMM ----------------
// C[M,N] = A[M,K] * B^T + epilogue, A rows contiguous in K (lda==Kd),
// Bsrc ALWAYS [N,K] row-major (ldb==Kd).
// EPI: 0 = +bias -> bf16 ; 1 = *scale -> f32 ; 2 = -> bf16 ; 3 = +bias -> f32
template<int EPI>
__global__ __launch_bounds__(256)
void wmma_gemm(const __bf16* __restrict__ A, const __bf16* __restrict__ Bsrc,
               const float* __restrict__ bias, float scale,
               float* __restrict__ outF, __bf16* __restrict__ outB,
               int M, int N, int Kd,
               long long sA, long long sB, long long sO)
{
  __shared__ __bf16 As[2][BMD][BKP];   // double-buffered
  __shared__ __bf16 Bs[2][BND][BKP];

  const int tid    = threadIdx.x;
  const int wave   = tid >> 5;          // wave32: 8 waves
  const int lane   = tid & 31;
  const int wm     = wave >> 1;         // 4 waves along M (32 rows each)
  const int wn     = wave & 1;          // 2 waves along N (64 cols each)
  const int blockM = blockIdx.y * BMD;
  const int blockN = blockIdx.x * BND;
  const int z      = blockIdx.z;

  A    += (long long)z * sA;
  Bsrc += (long long)z * sB;

  v8f zero = {};
  v8f acc[2][4] = {{zero, zero, zero, zero}, {zero, zero, zero, zero}};

  // prologue: async-stage tile 0
  async_tile_128x32(A,    Kd, blockM, 0, As[0], tid);
  async_tile_128x32(Bsrc, Kd, blockN, 0, Bs[0], tid);
  async_wait0();
  __syncthreads();

  int cur = 0;
  for (int kt = 0; kt < Kd; kt += BKD) {
    const bool more = (kt + BKD) < Kd;
    if (more) {  // overlap next-tile DMA with current-tile math
      async_tile_128x32(A,    Kd, blockM, kt + BKD, As[cur ^ 1], tid);
      async_tile_128x32(Bsrc, Kd, blockN, kt + BKD, Bs[cur ^ 1], tid);
    }

    const __bf16* as = &As[cur][0][0];
    const __bf16* bs = &Bs[cur][0][0];
    v16bf a0 = ld_frag(as, wm * 32,      lane);
    v16bf a1 = ld_frag(as, wm * 32 + 16, lane);
    v16bf b0 = ld_frag(bs, wn * 64,      lane);
    v16bf b1 = ld_frag(bs, wn * 64 + 16, lane);
    v16bf b2 = ld_frag(bs, wn * 64 + 32, lane);
    v16bf b3 = ld_frag(bs, wn * 64 + 48, lane);

    acc[0][0] = __builtin_amdgcn_wmma_f32_16x16x32_bf16(false, a0, false, b0, (short)0, acc[0][0], false, false);
    acc[0][1] = __builtin_amdgcn_wmma_f32_16x16x32_bf16(false, a0, false, b1, (short)0, acc[0][1], false, false);
    acc[0][2] = __builtin_amdgcn_wmma_f32_16x16x32_bf16(false, a0, false, b2, (short)0, acc[0][2], false, false);
    acc[0][3] = __builtin_amdgcn_wmma_f32_16x16x32_bf16(false, a0, false, b3, (short)0, acc[0][3], false, false);
    acc[1][0] = __builtin_amdgcn_wmma_f32_16x16x32_bf16(false, a1, false, b0, (short)0, acc[1][0], false, false);
    acc[1][1] = __builtin_amdgcn_wmma_f32_16x16x32_bf16(false, a1, false, b1, (short)0, acc[1][1], false, false);
    acc[1][2] = __builtin_amdgcn_wmma_f32_16x16x32_bf16(false, a1, false, b2, (short)0, acc[1][2], false, false);
    acc[1][3] = __builtin_amdgcn_wmma_f32_16x16x32_bf16(false, a1, false, b3, (short)0, acc[1][3], false, false);

    if (more) {
      async_wait0();       // own DMA for next tile done
      __syncthreads();     // everyone done reading cur & writing cur^1
      cur ^= 1;
    }
  }

  // ---- epilogue; C/D layout: VGPR r -> M = r + 8*(lane>>4), N = lane&15 ----
  const int l15  = lane & 15;
  const int half = (lane >> 4) & 1;
  float*  oF = outF + (long long)z * sO;
  __bf16* oB = outB + (long long)z * sO;

#pragma unroll
  for (int ti = 0; ti < 2; ++ti)
#pragma unroll
    for (int tj = 0; tj < 4; ++tj)
#pragma unroll
      for (int r = 0; r < 8; ++r) {
        int gm = blockM + wm * 32 + ti * 16 + r + 8 * half;
        int gn = blockN + wn * 64 + tj * 16 + l15;
        float val = acc[ti][tj][r];
        if (EPI == 0 || EPI == 3) val += bias[gn];
        if (EPI == 1)             val *= scale;
        if (EPI == 0 || EPI == 2) oB[(long long)gm * N + gn] = (__bf16)val;
        else                      oF[(long long)gm * N + gn] = val;
      }
}

// ---------------- bf16 matrix transpose: out[C,R] = in[R,C]^T ----------------
__global__ __launch_bounds__(256)
void transpose_bf16(const __bf16* __restrict__ in, __bf16* __restrict__ out,
                    int R, int C, long long sIn, long long sOut) {
  __shared__ __bf16 t[64][65];
  const int z  = blockIdx.z;
  in  += (long long)z * sIn;
  out += (long long)z * sOut;
  const int r0 = blockIdx.y * 64, c0 = blockIdx.x * 64;
  const int tid = threadIdx.x;
#pragma unroll
  for (int i = 0; i < 2; ++i) {
    int id = tid + 256 * i;
    int row = id >> 3;              // 8 x b128 per 64-elem row
    int c8  = (id & 7) * 8;
    bf16x8 v = *(const bf16x8*)(in + (long long)(r0 + row) * C + c0 + c8);
#pragma unroll
    for (int j = 0; j < 8; ++j) t[row][c8 + j] = v[j];
  }
  __syncthreads();
#pragma unroll
  for (int i = 0; i < 2; ++i) {
    int id = tid + 256 * i;
    int row = id >> 3;
    int c8  = (id & 7) * 8;
    bf16x8 o;
#pragma unroll
    for (int j = 0; j < 8; ++j) o[j] = t[c8 + j][row];
    *(bf16x8*)(out + (long long)(c0 + row) * R + r0 + c8) = o;
  }
}

// ---------------- row softmax: 2048 fp32 -> 2048 bf16 ----------------
__global__ __launch_bounds__(256)
void softmax_rows(const float* __restrict__ Sc, __bf16* __restrict__ P, int n) {
  const int row  = blockIdx.x;
  const int tid  = threadIdx.x;
  const int lane = tid & 31;
  const int warp = tid >> 5;
  const float* x = Sc + (long long)row * n;
  __bf16*      y = P  + (long long)row * n;
  __shared__ float redmax[8];
  __shared__ float redsum[8];

  float v[8];
  const int base = tid * 8;
  const float4* xv = (const float4*)(x + base);
  float4 a = xv[0], b = xv[1];
  v[0]=a.x; v[1]=a.y; v[2]=a.z; v[3]=a.w;
  v[4]=b.x; v[5]=b.y; v[6]=b.z; v[7]=b.w;

  float m = v[0];
#pragma unroll
  for (int i = 1; i < 8; ++i) m = fmaxf(m, v[i]);
#pragma unroll
  for (int o = 16; o > 0; o >>= 1) m = fmaxf(m, __shfl_xor(m, o, 32));
  if (lane == 0) redmax[warp] = m;
  __syncthreads();
  if (warp == 0) {
    float t = (lane < 8) ? redmax[lane] : -3.4e38f;
#pragma unroll
    for (int o = 4; o > 0; o >>= 1) t = fmaxf(t, __shfl_xor(t, o, 32));
    if (lane == 0) redmax[0] = t;
  }
  __syncthreads();
  m = redmax[0];

  float s = 0.f;
#pragma unroll
  for (int i = 0; i < 8; ++i) { v[i] = __expf(v[i] - m); s += v[i]; }
#pragma unroll
  for (int o = 16; o > 0; o >>= 1) s += __shfl_xor(s, o, 32);
  if (lane == 0) redsum[warp] = s;
  __syncthreads();
  if (warp == 0) {
    float t = (lane < 8) ? redsum[lane] : 0.f;
#pragma unroll
    for (int o = 4; o > 0; o >>= 1) t += __shfl_xor(t, o, 32);
    if (lane == 0) redsum[0] = t;
  }
  __syncthreads();
  const float inv = 1.0f / redsum[0];
#pragma unroll
  for (int i = 0; i < 8; ++i) y[base + i] = (__bf16)(v[i] * inv);
}

// ---------------- host launcher ----------------
extern "C" void kernel_launch(void* const* d_in, const int* in_sizes, int n_in,
                              void* d_out, int out_size, void* d_ws, size_t ws_size,
                              hipStream_t stream) {
  (void)in_sizes; (void)n_in; (void)out_size; (void)ws_size;
  const int Bb = 4, S = 2048, Dd = 1024, E = 2048;     // E = 2*D
  const long long MS = (long long)Bb * S;               // 8192 rows

  const float* hs = (const float*)d_in[0];
  const float* Wq = (const float*)d_in[1];
  const float* bq = (const float*)d_in[2];
  const float* Wk = (const float*)d_in[3];
  const float* bk = (const float*)d_in[4];
  const float* Wv = (const float*)d_in[5];
  const float* bv = (const float*)d_in[6];
  const float* Wo = (const float*)d_in[7];
  const float* bo = (const float*)d_in[8];
  float* out = (float*)d_out;

  char* ws = (char*)d_ws;
  size_t off = 0;
  auto carve = [&](size_t bytes) -> void* {
    void* p = ws + off;
    off = (off + bytes + 255) & ~(size_t)255;
    return p;
  };
  __bf16* hsb = (__bf16*)carve((size_t)MS * Dd * 2);
  __bf16* wqb = (__bf16*)carve((size_t)E * Dd * 2);
  __bf16* wkb = (__bf16*)carve((size_t)E * Dd * 2);
  __bf16* wvb = (__bf16*)carve((size_t)E * Dd * 2);
  __bf16* wob = (__bf16*)carve((size_t)Dd * E * 2);
  __bf16* Qb  = (__bf16*)carve((size_t)MS * E * 2);
  __bf16* Kb  = (__bf16*)carve((size_t)MS * E * 2);
  __bf16* Vb  = (__bf16*)carve((size_t)MS * E * 2);
  __bf16* Vt  = (__bf16*)carve((size_t)MS * E * 2);   // per-batch [E,S]
  float*  Scr = (float*)carve((size_t)Bb * S * S * 4);
  __bf16* Pb  = (__bf16*)carve((size_t)Bb * S * S * 2);
  __bf16* Ob  = (__bf16*)carve((size_t)MS * E * 2);

  // 1) fp32 -> bf16 conversions
  f32_to_bf16_kernel<<<4096, 256, 0, stream>>>(hs, hsb, MS * Dd);
  f32_to_bf16_kernel<<<2048, 256, 0, stream>>>(Wq, wqb, (long long)E * Dd);
  f32_to_bf16_kernel<<<2048, 256, 0, stream>>>(Wk, wkb, (long long)E * Dd);
  f32_to_bf16_kernel<<<2048, 256, 0, stream>>>(Wv, wvb, (long long)E * Dd);
  f32_to_bf16_kernel<<<2048, 256, 0, stream>>>(Wo, wob, (long long)Dd * E);

  // 2) Q/K/V projections: [8192,2048] = hs[8192,1024] @ W[2048,1024]^T + b
  dim3 gQKV(E / BND, MS / BMD, 1);
  wmma_gemm<0><<<gQKV, 256, 0, stream>>>(hsb, wqb, bq, 1.f, out, Qb,
                                         (int)MS, E, Dd, 0, 0, 0);
  wmma_gemm<0><<<gQKV, 256, 0, stream>>>(hsb, wkb, bk, 1.f, out, Kb,
                                         (int)MS, E, Dd, 0, 0, 0);
  wmma_gemm<0><<<gQKV, 256, 0, stream>>>(hsb, wvb, bv, 1.f, out, Vb,
                                         (int)MS, E, Dd, 0, 0, 0);

  // 3) Vt[b] = V[b]^T  ([S,E] -> [E,S]) so the P@V GEMM can use the [N,K] path
  dim3 gT(E / 64, S / 64, Bb);
  transpose_bf16<<<gT, 256, 0, stream>>>(Vb, Vt, S, E,
                                         (long long)S * E, (long long)S * E);

  // 4) scores[b] = (Q[b] @ K[b]^T) / sqrt(D)   (complex real part == full-dim dot)
  dim3 gSc(S / BND, S / BMD, Bb);
  wmma_gemm<1><<<gSc, 256, 0, stream>>>(Qb, Kb, bq, 1.0f / 32.0f, Scr, Qb,
                                        S, S, E,
                                        (long long)S * E, (long long)S * E,
                                        (long long)S * S);

  // 5) row softmax (fp32 math, bf16 probabilities)
  softmax_rows<<<Bb * S, 256, 0, stream>>>(Scr, Pb, S);

  // 6) O[b] = P[b] @ V[b] = P[b] @ (Vt[b])^T  (interleaved complex output for free)
  dim3 gPV(E / BND, S / BMD, Bb);
  wmma_gemm<2><<<gPV, 256, 0, stream>>>(Pb, Vt, bq, 1.f, Scr, Ob,
                                        S, E, S,
                                        (long long)S * S, (long long)S * E,
                                        (long long)S * E);

  // 7) out = O @ Wo^T + bo  -> fp32 result
  dim3 gO(Dd / BND, MS / BMD, 1);
  wmma_gemm<3><<<gO, 256, 0, stream>>>(Ob, wob, bo, 1.f, out, Ob,
                                       (int)MS, Dd, E, 0, 0, 0);
}
